// ALIGNN_45191645889141
// MI455X (gfx1250) — matrix operand order repaired and checked by reading, not compiled, not run on hardware
//
#include <hip/hip_runtime.h>

#define TPB 256

typedef __attribute__((ext_vector_type(16))) _Float16 v16h;
typedef __attribute__((ext_vector_type(8)))  _Float16 v8h;
typedef __attribute__((ext_vector_type(8)))  float    v8f;

__device__ inline v16h ld_frag(const _Float16* p0, const _Float16* p1) {
  v8h lo = *(const v8h*)p0;
  v8h hi = *(const v8h*)p1;
  return __builtin_shufflevector(lo, hi, 0,1,2,3,4,5,6,7,8,9,10,11,12,13,14,15);
}

// ---------------- weight pack: W[K,N] f32 -> Wt[N,Kp] f16, zero-padded K ----
__global__ __launch_bounds__(TPB)
void pack_w_k(const float* __restrict__ W, int K, int N, int Kp,
              _Float16* __restrict__ O) {
  size_t total = (size_t)N * Kp;
  size_t stride = (size_t)gridDim.x * blockDim.x;
  for (size_t i = (size_t)blockIdx.x * blockDim.x + threadIdx.x; i < total; i += stride) {
    int k = (int)(i % Kp);
    int n = (int)(i / Kp);
    O[i] = (k < K) ? (_Float16)W[(size_t)k * N + n] : (_Float16)0.0f;
  }
}

// ---------------- activation convert: A[M,K] f32 -> O[M,Kp] f16 -------------
__global__ __launch_bounds__(TPB)
void cvt_pad_k(const float* __restrict__ A, int M, int K, int Kp,
               _Float16* __restrict__ O) {
  size_t total = (size_t)M * Kp;
  size_t stride = (size_t)gridDim.x * blockDim.x;
  for (size_t i = (size_t)blockIdx.x * blockDim.x + threadIdx.x; i < total; i += stride) {
    int k = (int)(i % Kp);
    size_t r = i / Kp;
    O[i] = (k < K) ? (_Float16)A[r * K + k] : (_Float16)0.0f;
  }
}

// ---------------- WMMA GEMM: C[M,NC] = A[M,Kp](f16) @ Wt(NC,Kp)^T + bias ----
// Block: 256 threads = 8 waves; tile 128 rows x 64 cols.
// Weight tile (64 cols x Kp) staged in LDS via async DMA, then each wave
// computes 16 rows x 64 cols = 4 WMMA accumulators reusing its A fragment.
// EPI==1: fused edge-gate epilogue  C += gS[src[row]] + gD[dst[row]]
template<int EPI>
__global__ __launch_bounds__(TPB)
void wmma_gemm_k(const _Float16* __restrict__ A, const _Float16* __restrict__ Wt,
                 const float* __restrict__ bias, float* __restrict__ C,
                 int M, int Kp, int NC,
                 const float* __restrict__ gS, const float* __restrict__ gD,
                 const int* __restrict__ src, const int* __restrict__ dst) {
  __shared__ _Float16 sB[64 * 256];       // 32 KB max (Kp <= 256)

  const int lane = threadIdx.x & 31;
  const int wave = threadIdx.x >> 5;
  const int colbase = blockIdx.y * 64;

  // ---- stage weight tile: contiguous 64*Kp halfs -> LDS, async DMA ----
  {
    const unsigned tile16 = (unsigned)(64 * Kp) / 8;   // number of 16B chunks
    const _Float16* gB = Wt + (size_t)colbase * Kp;
    unsigned lbase = (unsigned)(uintptr_t)&sB[0];
    for (unsigned c16 = threadIdx.x; c16 < tile16; c16 += TPB) {
      unsigned laddr = lbase + c16 * 16;
      unsigned long long gaddr = (unsigned long long)(uintptr_t)(gB + (size_t)c16 * 8);
      asm volatile("global_load_async_to_lds_b128 %0, %1, off"
                   :: "v"(laddr), "v"(gaddr) : "memory");
    }
    asm volatile("s_wait_asynccnt 0x0" ::: "memory");
  }
  __syncthreads();

  const int row0 = blockIdx.x * 128 + wave * 16;
  if (row0 >= M) return;

  // A fragment source: row = row0 + lane%16 ; K run [klo,klo+16)
  int rA = row0 + (lane & 15);
  if (rA >= M) rA = M - 1;               // clamp; garbage rows never stored
  const int klo = (lane >> 4) * 16;
  const int kbB = (lane >> 4) * 16;      // B frag K run [kbB, kbB+16)
  const _Float16* Arow = A + (size_t)rA * Kp;

  v8f acc0 = {}, acc1 = {}, acc2 = {}, acc3 = {};
  const int cB = lane & 15;              // column within 16-col subtile
  for (int k = 0; k < Kp; k += 32) {
    v16h a = ld_frag(Arow + k + klo, Arow + k + klo + 8);
    const _Float16* s0 = &sB[(size_t)(cB +  0) * Kp + k + kbB];
    const _Float16* s1 = &sB[(size_t)(cB + 16) * Kp + k + kbB];
    const _Float16* s2 = &sB[(size_t)(cB + 32) * Kp + k + kbB];
    const _Float16* s3 = &sB[(size_t)(cB + 48) * Kp + k + kbB];
    v16h b0 = ld_frag(s0, s0 + 8);
    v16h b1 = ld_frag(s1, s1 + 8);
    v16h b2 = ld_frag(s2, s2 + 8);
    v16h b3 = ld_frag(s3, s3 + 8);
    acc0 = __builtin_amdgcn_wmma_f32_16x16x32_f16(false, a, false, b0, (short)0, acc0, false, false);
    acc1 = __builtin_amdgcn_wmma_f32_16x16x32_f16(false, a, false, b1, (short)0, acc1, false, false);
    acc2 = __builtin_amdgcn_wmma_f32_16x16x32_f16(false, a, false, b2, (short)0, acc2, false, false);
    acc3 = __builtin_amdgcn_wmma_f32_16x16x32_f16(false, a, false, b3, (short)0, acc3, false, false);
  }

  const int rbase = row0 + ((lane >> 4) * 8);
  v8f accs[4] = {acc0, acc1, acc2, acc3};
#pragma unroll
  for (int j = 0; j < 4; ++j) {
    const int n = colbase + j * 16 + cB;
    const float bv = bias[n];
#pragma unroll
    for (int r = 0; r < 8; ++r) {
      int mr = rbase + r;
      if (mr < M) {
        float v = accs[j][r] + bv;
        if (EPI == 1)
          v += gS[(size_t)src[mr] * 256 + n] + gD[(size_t)dst[mr] * 256 + n];
        C[(size_t)mr * NC + n] = v;
      }
    }
  }
}

// ---------------- per-column sum / sumsq for BatchNorm ----------------------
__global__ __launch_bounds__(TPB)
void colstats_k(const float* __restrict__ X, int M, int C, float* __restrict__ stats) {
  int c = threadIdx.x % C;
  int rstep = TPB / C;
  int rend = blockIdx.x * 128 + 128; if (rend > M) rend = M;
  float s = 0.f, sq = 0.f;
  for (int r = blockIdx.x * 128 + threadIdx.x / C; r < rend; r += rstep) {
    float v = X[(size_t)r * C + c];
    s += v; sq += v * v;
  }
  atomicAdd(&stats[c], s);
  atomicAdd(&stats[C + c], sq);
}

// ---------------- out = (RES ? R + silu(bn(T)) : silu(bn(T))) ---------------
template<int RES>
__global__ __launch_bounds__(TPB)
void bn_silu_k(const float* __restrict__ T, const float* __restrict__ R,
               const float* __restrict__ stats, const float* __restrict__ g,
               const float* __restrict__ bt, float invM,
               float* __restrict__ out, size_t total, int C) {
  size_t stride = (size_t)gridDim.x * blockDim.x;
  for (size_t i = (size_t)blockIdx.x * blockDim.x + threadIdx.x; i < total; i += stride) {
    int c = (int)(i % C);
    float mean = stats[c] * invM;
    float var  = stats[C + c] * invM - mean * mean;
    float y = (T[i] - mean) * rsqrtf(var + 1e-5f) * g[c] + bt[c];
    float s = y / (1.0f + __expf(-y));           // SiLU
    out[i] = (RES ? R[i] + s : s);
  }
}

// ---------------- edge gating scatter: num/den segment sums -----------------
__global__ __launch_bounds__(TPB)
void edge_scatter_k(const float* __restrict__ m, const float* __restrict__ Bh,
                    const int* __restrict__ src, const int* __restrict__ dst,
                    float* __restrict__ num, float* __restrict__ den, int E) {
  size_t total = (size_t)E * 256;
  size_t stride = (size_t)gridDim.x * blockDim.x;
  for (size_t i = (size_t)blockIdx.x * blockDim.x + threadIdx.x; i < total; i += stride) {
    size_t e = i >> 8; int c = (int)(i & 255);
    float sg = 1.0f / (1.0f + __expf(-m[i]));
    int sv = src[e], dv = dst[e];
    atomicAdd(&num[(size_t)dv * 256 + c], sg * Bh[(size_t)sv * 256 + c]);
    atomicAdd(&den[(size_t)dv * 256 + c], sg);
  }
}

// ---------------- u += num / (den + 1e-6) -----------------------------------
__global__ __launch_bounds__(TPB)
void gate_combine_k(float* __restrict__ u, const float* __restrict__ num,
                    const float* __restrict__ den, size_t total) {
  size_t stride = (size_t)gridDim.x * blockDim.x;
  for (size_t i = (size_t)blockIdx.x * blockDim.x + threadIdx.x; i < total; i += stride)
    u[i] += num[i] / (den[i] + 1e-6f);
}

// ---------------- avg-pool scatter ------------------------------------------
__global__ __launch_bounds__(TPB)
void pool_k(const float* __restrict__ h, const int* __restrict__ gid,
            float* __restrict__ pooled, float* __restrict__ cnt, int N) {
  size_t total = (size_t)N * 256;
  size_t stride = (size_t)gridDim.x * blockDim.x;
  for (size_t i = (size_t)blockIdx.x * blockDim.x + threadIdx.x; i < total; i += stride) {
    size_t r = i >> 8; int c = (int)(i & 255);
    int g = gid[r];
    atomicAdd(&pooled[(size_t)g * 256 + c], h[i]);
    if (c == 0) atomicAdd(&cnt[g], 1.0f);
  }
}

__global__ void readout_k(const float* __restrict__ pooled, const float* __restrict__ cnt,
                          const float* __restrict__ fW, const float* __restrict__ fb,
                          float* __restrict__ out, int B) {
  int b = blockIdx.x * blockDim.x + threadIdx.x;
  if (b >= B) return;
  float inv = 1.0f / fmaxf(cnt[b], 1.0f);
  float s = 0.f;
  for (int c = 0; c < 256; ++c) s += pooled[(size_t)b * 256 + c] * inv * fW[c];
  out[b] = s + fb[0];
}

static inline int gb(size_t total) {
  size_t b = (total + TPB - 1) / TPB;
  return (int)(b < 16384 ? b : 16384);
}

extern "C" void kernel_launch(void* const* d_in, const int* in_sizes, int n_in,
                              void* d_out, int out_size, void* d_ws, size_t ws_size,
                              hipStream_t stream) {
  const float* x    = (const float*)d_in[0];
  const float* erbf = (const float*)d_in[1];
  const float* ang  = (const float*)d_in[2];
  const int* src = (const int*)d_in[3];
  const int* dst = (const int*)d_in[4];
  const int* lgs = (const int*)d_in[5];
  const int* lgd = (const int*)d_in[6];
  const int* gid = (const int*)d_in[7];
  const float* atW=(const float*)d_in[8],  *atb=(const float*)d_in[9],
             *atg=(const float*)d_in[10], *atbt=(const float*)d_in[11];
  const float* e1W=(const float*)d_in[12], *e1b=(const float*)d_in[13],
             *e1g=(const float*)d_in[14], *e1bt=(const float*)d_in[15];
  const float* e2W=(const float*)d_in[16], *e2b=(const float*)d_in[17],
             *e2g=(const float*)d_in[18], *e2bt=(const float*)d_in[19];
  const float* t1W=(const float*)d_in[20], *t1b=(const float*)d_in[21],
             *t1g=(const float*)d_in[22], *t1bt=(const float*)d_in[23];
  const float* t2W=(const float*)d_in[24], *t2b=(const float*)d_in[25],
             *t2g=(const float*)d_in[26], *t2bt=(const float*)d_in[27];
  const float* cWp=(const float*)d_in[28], *cbp=(const float*)d_in[29],
             *cgp=(const float*)d_in[30], *cbtp=(const float*)d_in[31];
  const float* fW=(const float*)d_in[32], *fb=(const float*)d_in[33];
  float* out = (float*)d_out;

  const int N = in_sizes[0] / 92;
  const int E = in_sizes[1] / 80;
  const int T = in_sizes[2] / 40;
  const int B = out_size;

  // ---- workspace bump allocator ----
  char* wp = (char*)d_ws;
  auto alloc = [&](size_t bytes) -> void* {
    void* r = (void*)wp; wp += (bytes + 255) & ~(size_t)255; return r;
  };
  float* hbuf = (float*)alloc((size_t)N * 256 * 4);
  float* ybuf = (float*)alloc((size_t)E * 256 * 4);
  float* zbuf = (float*)alloc((size_t)T * 256 * 4);
  float* bufA = (float*)alloc((size_t)E * 256 * 4);   // src_gate out
  float* bufD = (float*)alloc((size_t)E * 256 * 4);   // dst_gate out
  float* bufB = (float*)alloc((size_t)E * 256 * 4);   // Bh
  float* ubuf = (float*)alloc((size_t)E * 256 * 4);   // x@W3+b3 (+h)
  float* mbuf = (float*)alloc((size_t)T * 256 * 4);   // edge message m
  float* numb = (float*)alloc((size_t)E * 256 * 4);
  float* denb = (float*)alloc((size_t)E * 256 * 4);
  _Float16* xbf = (_Float16*)alloc((size_t)E * 256 * 2);
  _Float16* ybf = (_Float16*)alloc((size_t)T * 256 * 2);
  float* stats  = (float*)alloc(512 * 4);
  float* pooled = (float*)alloc((size_t)B * 256 * 4);
  float* cntb   = (float*)alloc((size_t)B * 4);
  _Float16* wts = (_Float16*)alloc((size_t)(24576 + 6144 + 16384 + 4096 + 16384
                                            + 30 * 65536) * 2);
  _Float16* wt_at = wts;
  _Float16* wt_e1 = wt_at + 256 * 96;
  _Float16* wt_e2 = wt_e1 + 64 * 96;
  _Float16* wt_t1 = wt_e2 + 256 * 64;
  _Float16* wt_t2 = wt_t1 + 64 * 64;
  _Float16* wt_cv = wt_t2 + 256 * 64;

  // ---- pack all weights to f16 [N,Kp] ----
  pack_w_k<<<gb(256*96), TPB, 0, stream>>>(atW, 92, 256, 96, wt_at);
  pack_w_k<<<gb(64*96),  TPB, 0, stream>>>(e1W, 80, 64, 96, wt_e1);
  pack_w_k<<<gb(256*64), TPB, 0, stream>>>(e2W, 64, 256, 64, wt_e2);
  pack_w_k<<<gb(64*64),  TPB, 0, stream>>>(t1W, 40, 64, 64, wt_t1);
  pack_w_k<<<gb(256*64), TPB, 0, stream>>>(t2W, 64, 256, 64, wt_t2);
  for (int l = 0; l < 6; ++l)
    for (int i = 0; i < 5; ++i)
      pack_w_k<<<gb(65536), TPB, 0, stream>>>(cWp + ((size_t)l*5 + i) * 65536,
                                              256, 256, 256,
                                              wt_cv + ((size_t)l*5 + i) * 65536);

  auto cvt = [&](const float* A, int M, int K, int Kp, _Float16* O) {
    cvt_pad_k<<<gb((size_t)M * Kp), TPB, 0, stream>>>(A, M, K, Kp, O);
  };
  auto gemm = [&](const _Float16* A, const _Float16* Wt, const float* bias,
                  float* C, int M, int Kp, int NC) {
    dim3 g((M + 127) / 128, NC / 64);
    wmma_gemm_k<0><<<g, TPB, 0, stream>>>(A, Wt, bias, C, M, Kp, NC,
                                          nullptr, nullptr, nullptr, nullptr);
  };
  auto gemm_edge = [&](const _Float16* A, const _Float16* Wt, const float* bias,
                       float* C, int M, const float* gS, const float* gD,
                       const int* s, const int* d) {
    dim3 g((M + 127) / 128, 4);
    wmma_gemm_k<1><<<g, TPB, 0, stream>>>(A, Wt, bias, C, M, 256, 256, gS, gD, s, d);
  };
  auto bn = [&](const float* Tm, const float* Rm, const float* g, const float* b,
                int M, int C, float* o) {
    hipMemsetAsync(stats, 0, (size_t)2 * C * sizeof(float), stream);
    colstats_k<<<dim3((M + 127) / 128), TPB, 0, stream>>>(Tm, M, C, stats);
    size_t tot = (size_t)M * C;
    if (Rm) bn_silu_k<1><<<gb(tot), TPB, 0, stream>>>(Tm, Rm, stats, g, b,
                                                      1.0f / (float)M, o, tot, C);
    else    bn_silu_k<0><<<gb(tot), TPB, 0, stream>>>(Tm, nullptr, stats, g, b,
                                                      1.0f / (float)M, o, tot, C);
  };

  // ---- embeddings ----
  cvt(x, N, 92, 96, xbf);
  gemm(xbf, wt_at, atb, ubuf, N, 96, 256);
  bn(ubuf, nullptr, atg, atbt, N, 256, hbuf);

  cvt(erbf, E, 80, 96, xbf);
  gemm(xbf, wt_e1, e1b, ubuf, E, 96, 64);
  bn(ubuf, nullptr, e1g, e1bt, E, 64, ubuf);
  cvt(ubuf, E, 64, 64, xbf);
  gemm(xbf, wt_e2, e2b, bufA, E, 64, 256);
  bn(bufA, nullptr, e2g, e2bt, E, 256, ybuf);

  cvt(ang, T, 40, 64, ybf);
  gemm(ybf, wt_t1, t1b, mbuf, T, 64, 64);
  bn(mbuf, nullptr, t1g, t1bt, T, 64, mbuf);
  cvt(mbuf, T, 64, 64, ybf);
  gemm(ybf, wt_t2, t2b, mbuf, T, 64, 256);
  bn(mbuf, nullptr, t2g, t2bt, T, 256, zbuf);

  // ---- EdgeGatedGraphConv (in-place update of X [n,256] and Y [e,256]) ----
  auto egg = [&](int l, float* X, int n, float* Y, int e,
                 const int* sp, const int* dp) {
    const _Float16* w = wt_cv + (size_t)l * 5 * 65536;
    const float* bb = cbp + (size_t)l * 5 * 256;
    cvt(X, n, 256, 256, xbf);
    cvt(Y, e, 256, 256, ybf);
    gemm(xbf, w + 0 * 65536, bb + 0 * 256, bufA, n, 256, 256);   // src_gate
    gemm(xbf, w + 1 * 65536, bb + 1 * 256, bufD, n, 256, 256);   // dst_gate
    gemm(xbf, w + 4 * 65536, bb + 4 * 256, bufB, n, 256, 256);   // Bh
    gemm(xbf, w + 3 * 65536, bb + 3 * 256, ubuf, n, 256, 256);   // src_update
    gemm_edge(ybf, w + 2 * 65536, bb + 2 * 256, mbuf, e, bufA, bufD, sp, dp);
    hipMemsetAsync(numb, 0, (size_t)n * 256 * 4, stream);
    hipMemsetAsync(denb, 0, (size_t)n * 256 * 4, stream);
    edge_scatter_k<<<gb((size_t)e * 256), TPB, 0, stream>>>(mbuf, bufB, sp, dp,
                                                            numb, denb, e);
    gate_combine_k<<<gb((size_t)n * 256), TPB, 0, stream>>>(ubuf, numb, denb,
                                                            (size_t)n * 256);
    bn(ubuf, X, cgp + (size_t)(l*2+0)*256, cbtp + (size_t)(l*2+0)*256, n, 256, X);
    bn(mbuf, Y, cgp + (size_t)(l*2+1)*256, cbtp + (size_t)(l*2+1)*256, e, 256, Y);
  };

  egg(0, hbuf, N, ybuf, E, src, dst);
  egg(1, ybuf, E, zbuf, T, lgs, lgd);
  egg(2, hbuf, N, ybuf, E, src, dst);
  egg(3, ybuf, E, zbuf, T, lgs, lgd);
  egg(4, hbuf, N, ybuf, E, src, dst);
  egg(5, hbuf, N, ybuf, E, src, dst);

  // ---- pooling + readout ----
  hipMemsetAsync(pooled, 0, (size_t)B * 256 * 4, stream);
  hipMemsetAsync(cntb, 0, (size_t)B * 4, stream);
  pool_k<<<gb((size_t)N * 256), TPB, 0, stream>>>(hbuf, gid, pooled, cntb, N);
  readout_k<<<(B + 63) / 64, 64, 0, stream>>>(pooled, cntb, fW, fb, out, B);
}
